// MultiQueryAttention_6992206758372
// MI455X (gfx1250) — matrix operand order repaired and checked by reading, not compiled, not run on hardware
//
#include <hip/hip_runtime.h>
#include <math.h>

typedef __attribute__((ext_vector_type(16))) __bf16 v16bf;
typedef __attribute__((ext_vector_type(8)))  __bf16 v8bf;
typedef __attribute__((ext_vector_type(8)))  float  v8f;
typedef __attribute__((ext_vector_type(4)))  int    v4i;

// address-space-qualified 16B vector pointers for the async global->LDS builtin
typedef __attribute__((address_space(1))) v4i gv4i;  // global
typedef __attribute__((address_space(3))) v4i lv4i;  // LDS

#define S_LEN 4096
#define DIM   1024
#define HEADS 16
#define HD    64

// ---------------------------------------------------------------------------
// CDNA5 async global->LDS copy (ASYNCcnt-tracked), with safe sync fallback.
// ---------------------------------------------------------------------------
#if defined(__has_builtin)
#if __has_builtin(__builtin_amdgcn_global_load_async_to_lds_b128)
#define ASYNC_LDS 1
#endif
#if __has_builtin(__builtin_amdgcn_s_wait_asynccnt)
#define HAVE_WAIT_ASYNC_BUILTIN 1
#endif
#endif
#ifndef ASYNC_LDS
#define ASYNC_LDS 0
#endif
#ifndef HAVE_WAIT_ASYNC_BUILTIN
#define HAVE_WAIT_ASYNC_BUILTIN 0
#endif

__device__ __forceinline__ void async_copy16(const void* g, void* l) {
#if ASYNC_LDS
  __builtin_amdgcn_global_load_async_to_lds_b128((gv4i*)g, (lv4i*)l, 0, 0);
#else
  *(v8bf*)l = *(const v8bf*)g;
#endif
}

template <int N>
__device__ __forceinline__ void wait_async() {
#if ASYNC_LDS
#if HAVE_WAIT_ASYNC_BUILTIN
  __builtin_amdgcn_s_wait_asynccnt(N);
#else
  asm volatile("s_wait_asynccnt %0" ::"n"(N) : "memory");
#endif
#endif
}

// ---------------------------------------------------------------------------
// fp32 -> bf16 conversion kernels
// ---------------------------------------------------------------------------
__global__ void cvt_bf16(const float* __restrict__ src, __bf16* __restrict__ dst, int n) {
  int i = blockIdx.x * blockDim.x + threadIdx.x;
  if (i < n) dst[i] = (__bf16)src[i];
}

// src [R][C] f32  ->  dst [C][R] bf16   (weight transpose so B-operand is K-contiguous)
__global__ void cvt_bf16_T(const float* __restrict__ src, __bf16* __restrict__ dst, int R, int C) {
  int i = blockIdx.x * blockDim.x + threadIdx.x;
  if (i < R * C) {
    int r = i / C, c = i % C;
    dst[(size_t)c * R + r] = (__bf16)src[i];
  }
}

// Build a 16-bit A-fragment (16xK=32 slice) from two contiguous 8-half chunks.
// Layout (contiguous, k = 16*hi + s): lane<16: K = 0..15 ; lane>=16: K = 16..31.
__device__ __forceinline__ v16bf make_afrag(const __bf16* lo_p, const __bf16* hi_p) {
  v8bf lo = *(const v8bf*)lo_p;
  v8bf hi = *(const v8bf*)hi_p;
  v16bf a;
#pragma unroll
  for (int i = 0; i < 8; ++i) { a[i] = lo[i]; a[i + 8] = hi[i]; }
  return a;
}

// ---------------------------------------------------------------------------
// Tiled bf16 WMMA GEMM:  C[M][N] = A[M][K] * B[K][N],  B given as Bt[N][K].
// Block tile 128x64, 256 threads = 8 waves, each wave does 16 rows x 64 cols.
// Double-buffered LDS fed by async global->LDS copies (3 ops / thread / tile).
// EPI: 0 = f32 row-major, 1 = bf16 row-major, 2 = bf16 transposed [N][M].
// ---------------------------------------------------------------------------
template <int EPI>
__global__ __launch_bounds__(256) void gemm_wmma(
    const __bf16* __restrict__ A,   // [M][K] row-major
    const __bf16* __restrict__ Bt,  // [N][K] row-major (= B transposed)
    void* __restrict__ Cv, int M, int N, int K)
{
  __shared__ __bf16 As[2][128 * 32];
  __shared__ __bf16 Bs[2][64 * 32];

  const int tid  = threadIdx.x;
  const int wave = tid >> 5;
  const int lane = tid & 31;
  const int l15  = lane & 15;
  const int hi   = lane >> 4;
  const int m0   = blockIdx.x * 128;
  const int n0   = blockIdx.y * 64;

  const int arow = tid >> 1, ako = (tid & 1) * 16;
  const int brow = tid >> 2, bko = (tid & 3) * 8;

  auto issue_tile = [&](int k0, int buf) {
    const __bf16* asrc = &A[(size_t)(m0 + arow) * K + k0 + ako];
    async_copy16(asrc,     &As[buf][arow * 32 + ako]);
    async_copy16(asrc + 8, &As[buf][arow * 32 + ako + 8]);
    async_copy16(&Bt[(size_t)(n0 + brow) * K + k0 + bko], &Bs[buf][brow * 32 + bko]);
  };

  v8f acc[4] = {};
  issue_tile(0, 0);

  for (int k0 = 0, it = 0; k0 < K; k0 += 32, ++it) {
    const int buf = it & 1;
    const bool has_next = (k0 + 32 < K);
    if (has_next) {
      issue_tile(k0 + 32, buf ^ 1);
      wait_async<3>();  // in-order completion => tile `it` has landed
    } else {
      wait_async<0>();
    }
    __syncthreads();

    const int r = wave * 16 + l15;
    v16bf a = make_afrag(&As[buf][r * 32 + hi * 16], &As[buf][r * 32 + hi * 16 + 8]);

#pragma unroll
    for (int n = 0; n < 4; ++n) {
      // B-fragment: lane = column, 16 contiguous K values per half-wave
      v16bf b = *(const v16bf*)&Bs[buf][(n * 16 + l15) * 32 + hi * 16];
      acc[n] = __builtin_amdgcn_wmma_f32_16x16x32_bf16(
          false, a, false, b, (short)0, acc[n], false, false);
    }
    __syncthreads();  // protect buf^1 (written next iteration) until all reads done
  }

  // Epilogue. C/D layout: lane = col (l15), VGPR v = row v + 8*hi.
  if (EPI == 0) {
    float* C = (float*)Cv;
#pragma unroll
    for (int n = 0; n < 4; ++n)
#pragma unroll
      for (int v = 0; v < 8; ++v)
        C[(size_t)(m0 + wave * 16 + v + 8 * hi) * N + n0 + n * 16 + l15] = acc[n][v];
  } else if (EPI == 1) {
    __bf16* C = (__bf16*)Cv;
#pragma unroll
    for (int n = 0; n < 4; ++n)
#pragma unroll
      for (int v = 0; v < 8; ++v)
        C[(size_t)(m0 + wave * 16 + v + 8 * hi) * N + n0 + n * 16 + l15] =
            (__bf16)acc[n][v];
  } else {
    __bf16* Ct = (__bf16*)Cv;  // [N][M]
#pragma unroll
    for (int n = 0; n < 4; ++n) {
      v8bf pk;
#pragma unroll
      for (int v = 0; v < 8; ++v) pk[v] = (__bf16)acc[n][v];
      *(v8bf*)&Ct[(size_t)(n0 + n * 16 + l15) * M + m0 + wave * 16 + 8 * hi] = pk;
    }
  }
}

// ---------------------------------------------------------------------------
// Flash-style MQA attention. 4 waves / block; each wave owns one (head,
// 16-query) tile and iterates over keys in blocks of 32 (= WMMA K for PV).
// Online softmax, fp32 accumulation, bf16 matmul operands.
// K fragments are software-pipelined one key-block ahead; mask/V loads are
// issued at the top of the iteration so the S-WMMA + softmax window hides
// their latency. Row-sums come from a P @ ones WMMA (D-layout == lsum layout).
//   Q  : [S][DIM] bf16 ; Kb : [S][64] bf16 ; Vt : [64][S] bf16 ; mask f32
//   O  : [S][DIM] bf16 (row-major, feeds the final Wo GEMM as A)
// ---------------------------------------------------------------------------
__global__ __launch_bounds__(128) void flash_mqa(
    const __bf16* __restrict__ Q, const __bf16* __restrict__ Kb,
    const __bf16* __restrict__ Vt, const float* __restrict__ mask,
    __bf16* __restrict__ O)
{
  __shared__ __bf16 P[4][16 * 32];  // per-wave probability tile (D->A layout bounce)

  const int w    = threadIdx.x >> 5;
  const int lane = threadIdx.x & 31;
  const int l15  = lane & 15;
  const int hi   = lane >> 4;
  const int q0   = (blockIdx.x * 4 + w) * 16;
  const int h    = blockIdx.y;
  const float scale = 0.125f;  // 64^-0.5
  __bf16* Pw = &P[w][0];

  // Q fragments: A-operand, rows = queries, K-dim = d (64 = 2 chunks of 32)
  const __bf16* qrow = Q + (size_t)(q0 + l15) * DIM + h * HD;
  v16bf qf[2];
#pragma unroll
  for (int dc = 0; dc < 2; ++dc)
    qf[dc] = make_afrag(qrow + dc * 32 + hi * 16, qrow + dc * 32 + hi * 16 + 8);

  // all-ones B fragment for the row-sum WMMA
  v16bf ones;
#pragma unroll
  for (int i = 0; i < 16; ++i) ones[i] = (__bf16)1.0f;

  float m[8], lsum[8];
  v8f o[4] = {};
#pragma unroll
  for (int v = 0; v < 8; ++v) { m[v] = -INFINITY; lsum[v] = 0.0f; }

  // prologue: K fragments for key-block 0
  v16bf kf[2][2];
#pragma unroll
  for (int t = 0; t < 2; ++t) {
    const __bf16* krow = Kb + (size_t)(t * 16 + l15) * HD + hi * 16;
    kf[t][0] = *(const v16bf*)(krow);
    kf[t][1] = *(const v16bf*)(krow + 32);
  }

  for (int kb = 0; kb < S_LEN; kb += 32) {
    const bool has_next = (kb + 32 < S_LEN);
    if (has_next)
      __builtin_prefetch(&mask[(size_t)(q0 + l15) * S_LEN + kb + 32 + hi * 16], 0, 3);

    // ---- issue this block's mask + V loads early (consumed after S/softmax)
    float mk[2][8];
#pragma unroll
    for (int t = 0; t < 2; ++t)
#pragma unroll
      for (int v = 0; v < 8; ++v)
        mk[t][v] = mask[(size_t)(q0 + v + 8 * hi) * S_LEN + kb + t * 16 + l15];

    v16bf vf[4];
#pragma unroll
    for (int n = 0; n < 4; ++n)
      vf[n] = *(const v16bf*)&Vt[(size_t)(n * 16 + l15) * S_LEN + kb + hi * 16];

    // ---- S = Q @ K^T from the pipelined kf; dc-outer alternates
    //      accumulators to dodge the WMMA->WMMA RAW hazard
    v8f s[2] = {};
#pragma unroll
    for (int dc = 0; dc < 2; ++dc)
#pragma unroll
      for (int t = 0; t < 2; ++t)
        s[t] = __builtin_amdgcn_wmma_f32_16x16x32_bf16(
            false, qf[dc], false, kf[t][dc], (short)0, s[t], false, false);

    // ---- kf is now dead: issue next block's K loads (hidden by softmax+PV)
    v16bf kn[2][2];
    if (has_next) {
#pragma unroll
      for (int t = 0; t < 2; ++t) {
        const __bf16* krow = Kb + (size_t)(kb + 32 + t * 16 + l15) * HD + hi * 16;
        kn[t][0] = *(const v16bf*)(krow);
        kn[t][1] = *(const v16bf*)(krow + 32);
      }
    }

    // ---- softmax: scale+mask, row max (shfl over 16-lane half), exp, stage P
    float corr[8];
#pragma unroll
    for (int v = 0; v < 8; ++v) {
      float s0v = s[0][v] * scale + mk[0][v];
      float s1v = s[1][v] * scale + mk[1][v];
      float rm = fmaxf(s0v, s1v);
      rm = fmaxf(rm, __shfl_xor(rm, 1));
      rm = fmaxf(rm, __shfl_xor(rm, 2));
      rm = fmaxf(rm, __shfl_xor(rm, 4));
      rm = fmaxf(rm, __shfl_xor(rm, 8));
      float mn = fmaxf(m[v], rm);
      corr[v] = __expf(m[v] - mn);
      m[v] = mn;
      // stage P row-major in LDS: row = query (v + 8*hi), col = key-in-block
      Pw[(v + 8 * hi) * 32 + l15]      = (__bf16)__expf(s0v - mn);
      Pw[(v + 8 * hi) * 32 + 16 + l15] = (__bf16)__expf(s1v - mn);
    }

    // rescale running O by per-row correction (overlaps the DS traffic)
#pragma unroll
    for (int n = 0; n < 4; ++n)
#pragma unroll
      for (int v = 0; v < 8; ++v) o[n][v] *= corr[v];

    // LDS RAW fence (same-wave DS ops are in-order; wait before VALU use)
    asm volatile("s_wait_dscnt 0" ::: "memory");

    // reload P as an A-fragment (rows = queries, K = 32 keys)
    v16bf pf = make_afrag(&Pw[l15 * 32 + hi * 16], &Pw[l15 * 32 + hi * 16 + 8]);

    // ---- row-sums via WMMA: (P @ ones)[r][*] = sum_k P[r][k]; the D-layout
    //      places row r's sum exactly where lsum[v] lives in this lane
    v8f rs = __builtin_amdgcn_wmma_f32_16x16x32_bf16(
        false, pf, false, ones, (short)0, v8f{}, false, false);

    // ---- O += P @ V (4 independent accumulators) ------------------------
#pragma unroll
    for (int n = 0; n < 4; ++n)
      o[n] = __builtin_amdgcn_wmma_f32_16x16x32_bf16(
          false, pf, false, vf[n], (short)0, o[n], false, false);

#pragma unroll
    for (int v = 0; v < 8; ++v) lsum[v] = lsum[v] * corr[v] + rs[v];

    // rotate pipelined K fragments
    if (has_next) {
#pragma unroll
      for (int t = 0; t < 2; ++t) {
        kf[t][0] = kn[t][0];
        kf[t][1] = kn[t][1];
      }
    }
  }

  // ---- finalize: divide by l, write bf16 row-major [S][DIM] --------------
#pragma unroll
  for (int v = 0; v < 8; ++v) {
    float inv = 1.0f / lsum[v];
    const size_t row = (size_t)(q0 + v + 8 * hi) * DIM + h * HD;
#pragma unroll
    for (int n = 0; n < 4; ++n)
      O[row + n * 16 + l15] = (__bf16)(o[n][v] * inv);
  }
}

// ---------------------------------------------------------------------------
// Launch
// ---------------------------------------------------------------------------
extern "C" void kernel_launch(void* const* d_in, const int* in_sizes, int n_in,
                              void* d_out, int out_size, void* d_ws, size_t ws_size,
                              hipStream_t stream) {
  const float* x    = (const float*)d_in[0];
  const float* mask = (const float*)d_in[1];
  const float* Wq   = (const float*)d_in[2];
  const float* Wk   = (const float*)d_in[3];
  const float* Wv   = (const float*)d_in[4];
  const float* Wo   = (const float*)d_in[5];
  float* out = (float*)d_out;

  // workspace carve-out (~30 MB total)
  char* p = (char*)d_ws;
  auto carve = [&](size_t bytes) {
    char* q = p;
    p += (bytes + 255) & ~(size_t)255;
    return q;
  };
  __bf16* xbf  = (__bf16*)carve((size_t)S_LEN * DIM * 2);
  __bf16* WqT  = (__bf16*)carve((size_t)DIM * DIM * 2);
  __bf16* WkT  = (__bf16*)carve((size_t)DIM * HD * 2);
  __bf16* WvT  = (__bf16*)carve((size_t)DIM * HD * 2);
  __bf16* WoT  = (__bf16*)carve((size_t)DIM * DIM * 2);
  __bf16* Qbf  = (__bf16*)carve((size_t)S_LEN * DIM * 2);
  __bf16* Kbf  = (__bf16*)carve((size_t)S_LEN * HD * 2);
  __bf16* Vtb  = (__bf16*)carve((size_t)HD * S_LEN * 2);
  __bf16* AObf = (__bf16*)carve((size_t)S_LEN * DIM * 2);
  (void)ws_size;

  // 1) precision conversion (+ weight transposes)
  {
    int n = S_LEN * DIM;
    cvt_bf16<<<(n + 255) / 256, 256, 0, stream>>>(x, xbf, n);
    n = DIM * DIM;
    cvt_bf16_T<<<(n + 255) / 256, 256, 0, stream>>>(Wq, WqT, DIM, DIM);
    cvt_bf16_T<<<(n + 255) / 256, 256, 0, stream>>>(Wo, WoT, DIM, DIM);
    n = DIM * HD;
    cvt_bf16_T<<<(n + 255) / 256, 256, 0, stream>>>(Wk, WkT, DIM, HD);
    cvt_bf16_T<<<(n + 255) / 256, 256, 0, stream>>>(Wv, WvT, DIM, HD);
  }

  // 2) projections
  gemm_wmma<1><<<dim3(S_LEN / 128, DIM / 64), 256, 0, stream>>>(xbf, WqT, Qbf, S_LEN, DIM, DIM);
  gemm_wmma<1><<<dim3(S_LEN / 128, 1),        256, 0, stream>>>(xbf, WkT, Kbf, S_LEN, HD, DIM);
  gemm_wmma<2><<<dim3(S_LEN / 128, 1),        256, 0, stream>>>(xbf, WvT, Vtb, S_LEN, HD, DIM);

  // 3) flash attention (4 query tiles per block, one per wave)
  flash_mqa<<<dim3(S_LEN / 64, HEADS), 128, 0, stream>>>(Qbf, Kbf, Vtb, mask, AObf);

  // 4) output projection (f32 result)
  gemm_wmma<0><<<dim3(S_LEN / 128, DIM / 64), 256, 0, stream>>>(AObf, WoT, out, S_LEN, DIM, DIM);
}
